// vggnet_35210141892769
// MI455X (gfx1250) — matrix operand order repaired and checked by reading, not compiled, not run on hardware
//
#include <hip/hip_runtime.h>
#include <hip/hip_bf16.h>
#include <stdint.h>

// ---------------------------------------------------------------------------
// Types
// ---------------------------------------------------------------------------
typedef __bf16 bf16_t;
typedef __attribute__((ext_vector_type(16))) __bf16 v16bf;
typedef __attribute__((ext_vector_type(8)))  __bf16 v8bf;
typedef __attribute__((ext_vector_type(8)))  float  v8f;
typedef __attribute__((ext_vector_type(4)))  uint32_t v4u;
typedef __attribute__((ext_vector_type(8)))  uint32_t v8u;

__device__ __forceinline__ float bf2f(bf16_t b) {
  unsigned short u = __builtin_bit_cast(unsigned short, b);
  unsigned v = ((unsigned)u) << 16;
  return __builtin_bit_cast(float, v);
}
__device__ __forceinline__ bf16_t f2bf(float f) {
  unsigned u = __builtin_bit_cast(unsigned, f);
  unsigned r = (u + 0x7FFFu + ((u >> 16) & 1u)) >> 16;
  return __builtin_bit_cast(bf16_t, (unsigned short)r);
}

#define BM 128
#define BN 128
#define BK 32

// ---------------------------------------------------------------------------
// Weight binarization: sign(w) -> +/-1 in bf16
// ---------------------------------------------------------------------------
__global__ void k_binarize(const float* __restrict__ w, bf16_t* __restrict__ o, int n) {
  int i = blockIdx.x * blockDim.x + threadIdx.x;
  if (i < n) {
    unsigned short s = (w[i] >= 0.0f) ? (unsigned short)0x3F80 : (unsigned short)0xBF80;
    o[i] = __builtin_bit_cast(bf16_t, s);
  }
}

// ---------------------------------------------------------------------------
// Stem: 7x7 s2 SAME conv (fp32 weights) + ReLU + 2x2 maxpool, fused.
// x: [32,224,224,3]  w: [7,7,3,96]  y: [32,56,56,96] f32
// ---------------------------------------------------------------------------
__global__ void k_stem_pool(const float* __restrict__ x, const float* __restrict__ w,
                            float* __restrict__ y, int total) {
  int i = blockIdx.x * blockDim.x + threadIdx.x;
  if (i >= total) return;
  int oc = i % 96; int t = i / 96;
  int pw = t % 56; t /= 56;
  int ph = t % 56; t /= 56;
  int n  = t;
  float best = 0.0f;  // relu floor
  for (int dy = 0; dy < 2; ++dy)
    for (int dx = 0; dx < 2; ++dx) {
      int oh = ph * 2 + dy, ow = pw * 2 + dx;
      float acc = 0.0f;
      for (int kh = 0; kh < 7; ++kh) {
        int ih = oh * 2 + kh - 2;
        if ((unsigned)ih >= 224u) continue;
        for (int kw = 0; kw < 7; ++kw) {
          int iw = ow * 2 + kw - 2;
          if ((unsigned)iw >= 224u) continue;
          const float* px = x + ((size_t)(n * 224 + ih) * 224 + iw) * 3;
          const float* pk = w + ((size_t)(kh * 7 + kw) * 3) * 96 + oc;
          acc = fmaf(px[0], pk[0],   acc);
          acc = fmaf(px[1], pk[96],  acc);
          acc = fmaf(px[2], pk[192], acc);
        }
      }
      best = fmaxf(best, fmaxf(acc, 0.0f));
    }
  y[i] = best;
}

// ---------------------------------------------------------------------------
// BN (inference) on f32 input -> bf16 output, with extra scalar scale
// bn layout: [4][C] = gamma, beta, mean, var
// ---------------------------------------------------------------------------
__global__ void k_bn_f32_to_bf16(const float* __restrict__ x, const float* __restrict__ bn,
                                 const float* __restrict__ scale,
                                 bf16_t* __restrict__ y, int total, int C) {
  int i = blockIdx.x * blockDim.x + threadIdx.x;
  if (i >= total) return;
  int c = i % C;
  float g = bn[c], b = bn[C + c], m = bn[2 * C + c], v = bn[3 * C + c];
  float r = (g * (x[i] - m) * rsqrtf(v + 1e-3f) + b) * scale[0];
  y[i] = f2bf(r);
}

// BN in-place on bf16 buffer
__global__ void k_bn_bf16(bf16_t* __restrict__ x, const float* __restrict__ bn,
                          int total, int C) {
  int i = blockIdx.x * blockDim.x + threadIdx.x;
  if (i >= total) return;
  int c = i % C;
  float g = bn[c], b = bn[C + c], m = bn[2 * C + c], v = bn[3 * C + c];
  float r = g * (bf2f(x[i]) - m) * rsqrtf(v + 1e-3f) + b;
  x[i] = f2bf(r);
}

// 2x2 maxpool on bf16 NHWC
__global__ void k_pool_bf16(const bf16_t* __restrict__ x, bf16_t* __restrict__ y,
                            int N, int H, int W, int C) {
  int OH = H >> 1, OW = W >> 1;
  int total = N * OH * OW * C;
  int i = blockIdx.x * blockDim.x + threadIdx.x;
  if (i >= total) return;
  int c = i % C; int t = i / C;
  int pw = t % OW; t /= OW;
  int ph = t % OH; int n = t / OH;
  float m = -3.0e38f;
  for (int dy = 0; dy < 2; ++dy)
    for (int dx = 0; dx < 2; ++dx) {
      size_t o = ((size_t)(n * H + ph * 2 + dy) * W + (pw * 2 + dx)) * C + c;
      m = fmaxf(m, bf2f(x[o]));
    }
  y[i] = f2bf(m);
}

// ---------------------------------------------------------------------------
// Binarized implicit-GEMM conv / dense via WMMA bf16 (wave32).
//   act : NHWC bf16  [Nb,H,W,Cin]       (H=W=1, KH=KW=1 -> plain GEMM)
//   wgt : [K, Cout] bf16 (HWIO binarized, K = KH*KW*Cin), stride-1 SAME conv
// Block: 256 thr = 8 waves as 4x2, wave tile 32x64, block tile 128x128, BK=32.
// Weight tiles DMA'd by Tensor Data Mover into LDS (wave 0 issues + waits
// tensorcnt); activation tiles gathered (im2col) with b128 loads into LDS.
// All im2col index math is K-loop-invariant or incrementally updated: no
// integer division inside the K loop.
// mode: 0 = ReLU+BN -> bf16 ; 1 = ReLU -> bf16 ; 2 = *scalu -> f32
// ---------------------------------------------------------------------------
__launch_bounds__(256)
__global__ void k_bgemm_wmma(const bf16_t* __restrict__ act,
                             const bf16_t* __restrict__ wgt,
                             bf16_t* __restrict__ out_bf,
                             float* __restrict__ out_f32,
                             const float* __restrict__ bn,
                             const float* __restrict__ scalu,
                             int Nb, int H, int W, int Cin,
                             int Cout, int KH, int KW, int mode) {
  const int M   = Nb * H * W;
  const int K   = KH * KW * Cin;
  const int pad = KH >> 1;

  __shared__ __attribute__((aligned(16))) bf16_t sA[BM][BK + 8];
  __shared__ __attribute__((aligned(16))) bf16_t sBs[BK * BN];   // TDM staging [k][n]
  __shared__ __attribute__((aligned(16))) bf16_t sB[BN][BK + 8]; // col-major   [n][k]

  const int tid  = threadIdx.x;
  const int m0   = blockIdx.y * BM;
  const int n0   = blockIdx.x * BN;
  const int wid  = tid >> 5;
  const int lane = tid & 31;
  const int wm   = wid & 3;    // 0..3 (M)
  const int wn   = wid >> 2;   // 0..1 (N)
  const int lrow = lane & 15;
  const int lhi  = lane >> 4;  // 0/1

  v8f acc[2][4] = {};

  // ---- K-loop-invariant im2col decode for this thread's A-tile slot ----
  const int ar = tid >> 1;             // row in tile: 0..127
  const int ak = (tid & 1) << 4;       // 0 or 16
  const int m  = m0 + ar;
  const bool mvalid = m < M;
  {
  }
  const int msafe = mvalid ? m : 0;
  const int ow = msafe % W;
  const int t_ = msafe / W;
  const int oh = t_ % H;
  const int nb = t_ / H;
  const size_t rowStride = (size_t)W * (size_t)Cin;   // elems per (n,h) row
  const size_t nbase     = (size_t)(nb * H) * rowStride;
  // incremental k -> (kh, kw, ci): starts at k = ak (< 32 <= Cin)
  int ci = ak, kw = 0, kh = 0;

  const uint32_t lds_b_addr = (uint32_t)(uintptr_t)(&sBs[0]);

  for (int k0 = 0; k0 < K; k0 += BK) {
    // ---- A tile (im2col gather): 2 threads per row, 16 bf16 each ----
    {
      const int ih = oh + kh - pad;
      const int iw = ow + kw - pad;
      v4u lo = {}, hi = {};
      if (mvalid && (unsigned)ih < (unsigned)H && (unsigned)iw < (unsigned)W) {
        const bf16_t* p = act + nbase + (size_t)ih * rowStride
                              + (size_t)iw * (size_t)Cin + (size_t)ci;
        lo = *(const v4u*)(p);
        hi = *(const v4u*)(p + 8);
      }
      *(v4u*)&sA[ar][ak]     = lo;
      *(v4u*)&sA[ar][ak + 8] = hi;
      // advance this thread's k by BK (no division; BK <= Cin always)
      ci += BK;
      if (ci >= Cin) {
        ci -= Cin;
        if (++kw == KW) { kw = 0; ++kh; }
      }
    }

    // ---- B tile: Tensor Data Mover DMA (wave 0 only; EXEC-independent) ----
    if (wid == 0) {
      uint64_t gaddr = (uint64_t)(uintptr_t)(wgt + ((size_t)k0 * Cout + n0));
      uint32_t d0 = 1u;                                   // count=1, normal mode
      uint32_t d1 = lds_b_addr;                           // lds_addr
      uint32_t d2 = (uint32_t)gaddr;                      // global_addr[31:0]
      uint32_t d3 = (uint32_t)((gaddr >> 32) & 0x01FFFFFFu) | (2u << 30); // hi, type=2
      uint64_t dim0 = (uint64_t)(uint32_t)Cout;           // tensor_dim0 (elems)
      uint64_t dim1 = (uint64_t)(uint32_t)K;              // tensor_dim1
      uint64_t str0 = (uint64_t)(uint32_t)Cout;           // tensor_dim0_stride
      uint32_t e0 = (1u << 16);                           // wg_mask=0, data_size=1 (2B)
      uint32_t e1 = (uint32_t)((dim0 & 0xFFFFu) << 16);   // [63:48] dim0.lo16
      uint32_t e2 = (uint32_t)(((dim0 >> 16) & 0xFFFFu) | ((dim1 & 0xFFFFu) << 16));
      uint32_t e3 = (uint32_t)(((dim1 >> 16) & 0xFFFFu) | ((uint32_t)BN << 16)); // tile_dim0
      uint32_t e4 = (uint32_t)BK;                         // tile_dim1=BK, tile_dim2=0
      uint32_t e5 = (uint32_t)str0;                       // stride0[31:0]
      uint32_t e6 = (uint32_t)((str0 >> 32) & 0xFFFFu);   // stride0[47:32], stride1.lo=0
      uint32_t e7 = 0u;
      v4u g0v = { d0, d1, d2, d3 };
      v8u g1v = { e0, e1, e2, e3, e4, e5, e6, e7 };
      asm volatile("tensor_load_to_lds %0, %1" :: "s"(g0v), "s"(g1v) : "memory");
      __builtin_amdgcn_s_wait_tensorcnt(0);
    }
    __syncthreads();

    // ---- transpose staging [k][n] -> sB [n][k] (LDS->LDS) ----
    {
      const int base = tid * 16;
#pragma unroll
      for (int j = 0; j < 16; ++j) {
        int idx = base + j;            // 0..4095
        int nn  = idx & (BN - 1);
        int kk2 = idx >> 7;            // /BN
        sB[nn][kk2] = sBs[idx];
      }
    }
    __syncthreads();

    if (k0 + BK < K)  // hint next weight tile toward L2 (global_prefetch_b8)
      __builtin_prefetch((const void*)(wgt + ((size_t)(k0 + BK) * Cout + n0)), 0, 1);

    // ---- WMMA: 2x4 16x16 tiles per wave (wave tile 32x64) ----
#pragma unroll
    for (int mi = 0; mi < 2; ++mi) {
      const int arow = wm * 32 + mi * 16 + lrow;
      v8bf alo = *(const v8bf*)&sA[arow][lhi * 8];        // K 0-7 / 8-15
      v8bf ahi = *(const v8bf*)&sA[arow][lhi * 8 + 16];   // K 16-23 / 24-31
      v16bf a = __builtin_shufflevector(alo, ahi, 0,1,2,3,4,5,6,7,8,9,10,11,12,13,14,15);
#pragma unroll
      for (int ni = 0; ni < 4; ++ni) {
        const int bcol = wn * 64 + ni * 16 + lrow;
        v8bf blo = *(const v8bf*)&sB[bcol][lhi * 8];
        v8bf bhi = *(const v8bf*)&sB[bcol][lhi * 8 + 16];
        v16bf b = __builtin_shufflevector(blo, bhi, 0,1,2,3,4,5,6,7,8,9,10,11,12,13,14,15);
        acc[mi][ni] = __builtin_amdgcn_wmma_f32_16x16x32_bf16(
            false, a, false, b, (short)0, acc[mi][ni], false, false);
      }
    }
    __syncthreads();
  }

  // ---- epilogue: ReLU / BN / scalu, bf16 or f32 store ----
  const float su = (mode == 2) ? scalu[0] : 1.0f;
#pragma unroll
  for (int ni = 0; ni < 4; ++ni) {
    const int col = n0 + wn * 64 + ni * 16 + lrow;
    float g = 1.0f, bb = 0.0f, mmn = 0.0f, vv = 1.0f;
    if (mode == 0 && col < Cout) {
      g   = bn[col];            bb = bn[Cout + col];
      mmn = bn[2 * Cout + col]; vv = bn[3 * Cout + col];
    }
    const float inv = rsqrtf(vv + 1e-3f);
#pragma unroll
    for (int mi = 0; mi < 2; ++mi) {
#pragma unroll
      for (int r = 0; r < 8; ++r) {
        const int row = m0 + wm * 32 + mi * 16 + lhi * 8 + r;
        if (row < M && col < Cout) {
          float x = acc[mi][ni][r];
          if (mode != 2) x = fmaxf(x, 0.0f);
          if (mode == 0) x = g * (x - mmn) * inv + bb;
          size_t o = (size_t)row * Cout + col;
          if (mode == 2) out_f32[o] = x * su;
          else           out_bf[o]  = f2bf(x);
        }
      }
    }
  }
}

// ---------------------------------------------------------------------------
// Orchestration
// ---------------------------------------------------------------------------
extern "C" void kernel_launch(void* const* d_in, const int* in_sizes, int n_in,
                              void* d_out, int out_size, void* d_ws, size_t ws_size,
                              hipStream_t stream) {
  (void)in_sizes; (void)n_in; (void)out_size; (void)ws_size;
  const float* x     = (const float*)d_in[0];
  const float* w1    = (const float*)d_in[1];
  const float* wsrc[12];
  for (int i = 0; i < 9; ++i) wsrc[i]     = (const float*)d_in[2 + i];   // w2..w10
  for (int i = 0; i < 3; ++i) wsrc[9 + i] = (const float*)d_in[11 + i];  // wd1..wd3
  const float* bn[12];
  for (int i = 0; i < 12; ++i) bn[i] = (const float*)d_in[14 + i];
  const float* scale = (const float*)d_in[26];
  const float* scalu = (const float*)d_in[27];
  float* out = (float*)d_out;

  // bump allocator over workspace
  char* ws = (char*)d_ws;
  size_t off = 0;
  auto alloc = [&](size_t bytes) -> void* {
    void* p = ws + off;
    off = (off + bytes + 255) & ~(size_t)255;
    return p;
  };

  const int STEM_N  = 32 * 56 * 56 * 96;         // 9,633,792
  const int ACT_MAX = 32 * 56 * 56 * 256;        // 25,690,112
  float*  stemF = (float*)alloc((size_t)STEM_N * 4);
  bf16_t* actA  = (bf16_t*)alloc((size_t)ACT_MAX * 2);
  bf16_t* actB  = (bf16_t*)alloc((size_t)ACT_MAX * 2);
  bf16_t* actC  = (bf16_t*)alloc((size_t)ACT_MAX * 2);

  const int wn_[12] = {
    9*96*256, 9*256*256, 9*256*256, 9*256*512,
    9*512*512, 9*512*512, 9*512*512, 9*512*512, 9*512*512,
    25088*4096, 4096*4096, 4096*1000 };
  bf16_t* wbin[12];
  for (int i = 0; i < 12; ++i) wbin[i] = (bf16_t*)alloc((size_t)wn_[i] * 2);

  // 1) binarize all 1-bit weights to +/-1 bf16
  for (int i = 0; i < 12; ++i)
    k_binarize<<<(wn_[i] + 255) / 256, 256, 0, stream>>>(wsrc[i], wbin[i], wn_[i]);

  // 2) fp32 stem: conv7x7 s2 + relu + pool, then BN1 * scale -> bf16
  k_stem_pool<<<(STEM_N + 255) / 256, 256, 0, stream>>>(x, w1, stemF, STEM_N);
  k_bn_f32_to_bf16<<<(STEM_N + 255) / 256, 256, 0, stream>>>(stemF, bn[0], scale,
                                                             actA, STEM_N, 96);

  auto bgemm = [&](const bf16_t* a, const bf16_t* w, bf16_t* ob, float* of,
                   const float* bnp, int Nb, int H, int W, int Cin,
                   int Cout, int KH, int KW, int mode) {
    int M = Nb * H * W;
    dim3 grid((Cout + BN - 1) / BN, (M + BM - 1) / BM);
    k_bgemm_wmma<<<grid, 256, 0, stream>>>(a, w, ob, of, bnp, scalu,
                                           Nb, H, W, Cin, Cout, KH, KW, mode);
  };
  auto pool = [&](const bf16_t* a, bf16_t* o, int Nb, int H, int W, int C) {
    int total = Nb * (H / 2) * (W / 2) * C;
    k_pool_bf16<<<(total + 255) / 256, 256, 0, stream>>>(a, o, Nb, H, W, C);
  };
  auto bnip = [&](bf16_t* a, const float* p, int total, int C) {
    k_bn_bf16<<<(total + 255) / 256, 256, 0, stream>>>(a, p, total, C);
  };

  // 3) binarized conv trunk (56x56 stage)
  bgemm(actA, wbin[0], actB, nullptr, bn[1], 32, 56, 56,  96, 256, 3, 3, 0);
  bgemm(actB, wbin[1], actA, nullptr, bn[2], 32, 56, 56, 256, 256, 3, 3, 0);
  bgemm(actA, wbin[2], actB, nullptr, nullptr, 32, 56, 56, 256, 256, 3, 3, 1);
  pool(actB, actC, 32, 56, 56, 256);
  bnip(actC, bn[3], 32 * 28 * 28 * 256, 256);
  // 28x28 stage
  bgemm(actC, wbin[3], actA, nullptr, bn[4], 32, 28, 28, 256, 512, 3, 3, 0);
  bgemm(actA, wbin[4], actB, nullptr, bn[5], 32, 28, 28, 512, 512, 3, 3, 0);
  bgemm(actB, wbin[5], actA, nullptr, nullptr, 32, 28, 28, 512, 512, 3, 3, 1);
  pool(actA, actC, 32, 28, 28, 512);
  bnip(actC, bn[6], 32 * 14 * 14 * 512, 512);
  // 14x14 stage
  bgemm(actC, wbin[6], actA, nullptr, bn[7], 32, 14, 14, 512, 512, 3, 3, 0);
  bgemm(actA, wbin[7], actB, nullptr, bn[8], 32, 14, 14, 512, 512, 3, 3, 0);
  bgemm(actB, wbin[8], actA, nullptr, nullptr, 32, 14, 14, 512, 512, 3, 3, 1);
  pool(actA, actC, 32, 14, 14, 512);
  bnip(actC, bn[9], 32 * 7 * 7 * 512, 512);

  // 4) binarized dense head (flatten = NHWC order; GEMM = 1x1 conv, H=W=1)
  bgemm(actC, wbin[9],  actA, nullptr, bn[10], 32, 1, 1, 25088, 4096, 1, 1, 0);
  bgemm(actA, wbin[10], actB, nullptr, bn[11], 32, 1, 1,  4096, 4096, 1, 1, 0);
  bgemm(actB, wbin[11], nullptr, out, nullptr, 32, 1, 1,  4096, 1000, 1, 1, 2);
}
